// MSA_59897613910312
// MI455X (gfx1250) — matrix operand order repaired and checked by reading, not compiled, not run on hardware
//
#include <hip/hip_runtime.h>
#include <hip/hip_bf16.h>
#include <math.h>

typedef __attribute__((ext_vector_type(16))) _Float16 v16h;
typedef __attribute__((ext_vector_type(8)))  _Float16 v8h;
typedef __attribute__((ext_vector_type(8)))  float    v8f;

union FragA { v16h f; v8h h[2]; };
struct KF { FragA a00, a01, a10, a11; };  // K fragments for one 32-key tile

constexpr int E  = 1024;   // embed dim
constexpr int H  = 16;     // heads
constexpr int D  = 64;     // head dim
constexpr int NB = 2;      // batch
constexpr int S  = 2048;   // query len
constexpr int T  = 2048;   // key len
constexpr int R  = NB * S; // 4096 projection rows

// 1/sqrt(D) * log2(e): logits come out of the QK WMMA already in log2 domain,
// so softmax needs only v_exp_f32 (native exp2), no per-element multiply.
#define QSCALE (0.125f * 1.4426950408889634f)

static __device__ __forceinline__ v8f wmma_f16(v16h a, v16h b, v8f c) {
  // v_wmma_f32_16x16x32_f16 : D = A(16x32) * B(32x16) + C(16x16 f32)
  return __builtin_amdgcn_wmma_f32_16x16x32_f16(false, a, false, b, (short)0, c,
                                                false, false);
}

// Raw min/max: single v_min_num/v_max_num without the NaN-canonicalization
// (v_max x,x,x) ops the compiler inserts for fminf/fmaxf on opaque inputs.
// Safe here: WMMA outputs / mask loads are never sNaN.
static __device__ __forceinline__ float fmin_raw(float a, float b) {
  float r; asm("v_min_num_f32 %0, %1, %2" : "=v"(r) : "v"(a), "v"(b)); return r;
}
static __device__ __forceinline__ float fmax_raw(float a, float b) {
  float r; asm("v_max_num_f32 %0, %1, %2" : "=v"(r) : "v"(a), "v"(b)); return r;
}

// ---------------------------------------------------------------------------
// Projection GEMM: out[r, c] = (sum_k A[r,k] * W[c,k] + bias[c]) * scale.
// One wave per 16x16 output tile. f32 loads, cvt to f16, WMMA f32 accumulate.
// TRANS=true stores transposed per batch: out[(n*E + c) * S + t]  (for V).
// ---------------------------------------------------------------------------
template <bool TRANS>
__global__ __launch_bounds__(256) void proj_kernel(const float* __restrict__ A,
                                                   const float* __restrict__ W,
                                                   const float* __restrict__ bias,
                                                   _Float16* __restrict__ out,
                                                   float scale) {
  const int lane = threadIdx.x & 31;
  const int warp = __builtin_amdgcn_readfirstlane(threadIdx.x >> 5);
  const int wave = blockIdx.x * 8 + warp;          // uniform
  const int ct   = wave & (E / 16 - 1);            // 64 column tiles
  const int rt   = wave >> 6;                      // 256 row tiles
  const int hi   = lane >> 4;                      // half-wave select
  const int ln   = lane & 15;

  const float* arow = A + (size_t)(rt * 16 + ln) * E;  // A-frag: row = ln
  const float* wrow = W + (size_t)(ct * 16 + ln) * E;  // B-frag: col = ln

  v8f acc = {0.f, 0.f, 0.f, 0.f, 0.f, 0.f, 0.f, 0.f};
  for (int k = 0; k < E; k += 32) {
    FragA a, b;
    // A layout: lane holds row ln, K = hi*8+{0..7} then 16+hi*8+{0..7}
    a.h[0] = __builtin_convertvector(*(const v8f*)(arow + k + hi * 8), v8h);
    a.h[1] = __builtin_convertvector(*(const v8f*)(arow + k + 16 + hi * 8), v8h);
    // B layout: lane holds col ln, K = hi*8+{0..7} then 16+hi*8+{0..7}
    b.h[0] = __builtin_convertvector(*(const v8f*)(wrow + k + hi * 8), v8h);
    b.h[1] = __builtin_convertvector(*(const v8f*)(wrow + k + 16 + hi * 8), v8h);
    acc = wmma_f16(a.f, b.f, acc);
  }

  const float bv = bias[ct * 16 + ln];
  if (TRANS) {
    // C layout: lane holds rows M = v + 8*hi (v=0..7 consecutive) of col ln
    // -> transposed store is one contiguous v8h per lane.
    const int rbase = rt * 16 + hi * 8;
    const int n     = rbase >> 11;       // / S
    const int tb    = rbase & (S - 1);
    v8h o;
#pragma unroll
    for (int v = 0; v < 8; ++v) o[v] = (_Float16)((acc[v] + bv) * scale);
    *(v8h*)(out + (size_t)(n * E + ct * 16 + ln) * S + tb) = o;
  } else {
#pragma unroll
    for (int v = 0; v < 8; ++v)
      out[(size_t)(rt * 16 + v + 8 * hi) * E + ct * 16 + ln] =
          (_Float16)((acc[v] + bv) * scale);
  }
}

// ---------------------------------------------------------------------------
// meanV[n*E + e] = (1/T) * sum_t Vproj[n,t,e]  (from transposed Vt; rows are
// contiguous). Used for the exact degenerate-row (all-masked) fixup.
// ---------------------------------------------------------------------------
__global__ __launch_bounds__(256) void meanv_kernel(const _Float16* __restrict__ Vt,
                                                    float* __restrict__ meanV) {
  const int idx = blockIdx.x * 256 + threadIdx.x;  // 0 .. NB*E-1
  const _Float16* row = Vt + (size_t)idx * T;
  float s = 0.f;
  for (int t = 0; t < T; t += 8) {
    v8h x = *(const v8h*)(row + t);
#pragma unroll
    for (int i = 0; i < 8; ++i) s += (float)x[i];
  }
  meanV[idx] = s * (1.0f / (float)T);
}

// maskf[t] = pad ? +BIG : -1e9 ; logits get min(x, maskf) -> exact -1e9 when
// masked (preserves the degenerate-row detector), x when allowed.
__global__ __launch_bounds__(256) void padmask_kernel(const int* __restrict__ pad,
                                                      float* __restrict__ maskf) {
  const int i = blockIdx.x * 256 + threadIdx.x;    // 0 .. NB*T-1
  maskf[i] = (pad[i] != 0) ? 3.0e38f : -1.0e9f;
}

// ---------------------------------------------------------------------------
// Flash attention, transposed-logit formulation: St = K * Q^T so each lane
// owns one query row's logits AND already holds the exact A-fragment pattern
// for the P*V WMMA (no LDS; 2 shuffles per tile for the softmax reductions).
// Log2-domain softmax (scale folded into Q). Double-buffered K fragments
// (unroll-2, no rotating copies). Only the single tail tile is causal.
// Fully pad-masked rows detected by m == -1e9 -> meanV.
// ---------------------------------------------------------------------------
__global__ __launch_bounds__(256) void attn_kernel(const _Float16* __restrict__ Qp,
                                                   const _Float16* __restrict__ Kp,
                                                   const _Float16* __restrict__ Vt,
                                                   const float* __restrict__ maskf,
                                                   const float* __restrict__ meanV,
                                                   float* __restrict__ out) {
  const int lane = threadIdx.x & 31;
  const int warp = __builtin_amdgcn_readfirstlane(threadIdx.x >> 5);
  const int wave = blockIdx.x * 8 + warp;          // uniform
  const int qt   = wave & (S / 16 - 1);            // 128 query tiles
  const int nh   = wave >> 7;
  const int h    = nh & (H - 1);
  const int n    = nh >> 4;
  const int hi   = lane >> 4;
  const int ln   = lane & 15;
  const int q0   = qt * 16;                        // uniform

  // Q as B-fragments of Q^T: col = query q0+ln, K = feats hi*16+{0..15}
  const _Float16* qrow = Qp + (size_t)(n * S + q0 + ln) * E + h * D;
  const v16h qb0 = *(const v16h*)(qrow + hi * 16);
  const v16h qb1 = *(const v16h*)(qrow + 32 + hi * 16);

  const v8f vzero = {0.f, 0.f, 0.f, 0.f, 0.f, 0.f, 0.f, 0.f};
  v8f acc0 = vzero, acc1 = vzero, acc2 = vzero, acc3 = vzero;
  float m = -3.0e38f;  // running row max (per lane = per query row)
  float l = 0.f;       // running row sum

  // Per-lane base pointers; j enters only as a uniform offset.
  const _Float16* krow_l = Kp + (size_t)(n * T + ln) * E + h * D;
  const _Float16* vcol_l = Vt + (size_t)(n * E + h * D + ln) * T + hi * 8;
  const float*    mrow_l = maskf + n * T + 8 * hi;
  const int       srow   = q0 + ln;   // this lane's query row
  // Tiles [j, j+32) with j+31 <= q0 never hit the causal mask. q0 is a
  // multiple of 16, so exactly ONE tail tile (at jfull) needs the compares.
  const int jfull  = (q0 + 1) & ~31;  // uniform
  const int ntiles = (jfull >> 5) + 1;

  auto load_ka = [&](int j) {
    KF f;
    const _Float16* kr0 = krow_l + (size_t)j * E;
    const _Float16* kr1 = kr0 + (size_t)16 * E;
    f.a00.h[0] = *(const v8h*)(kr0 + hi * 16);
    f.a00.h[1] = *(const v8h*)(kr0 + hi * 16 + 8);
    f.a01.h[0] = *(const v8h*)(kr0 + 32 + hi * 16);
    f.a01.h[1] = *(const v8h*)(kr0 + 32 + hi * 16 + 8);
    f.a10.h[0] = *(const v8h*)(kr1 + hi * 16);
    f.a10.h[1] = *(const v8h*)(kr1 + hi * 16 + 8);
    f.a11.h[0] = *(const v8h*)(kr1 + 32 + hi * 16);
    f.a11.h[1] = *(const v8h*)(kr1 + 32 + hi * 16 + 8);
    __builtin_prefetch((const void*)(kr0 + (size_t)32 * E), 0, 1);
    return f;
  };

  auto tile = [&](int j, const KF& kf, int causal) {
    // St tiles: element v of st0 = log2-logit(key j+v+8*hi, query q0+ln)
    v8f st0 = vzero, st1 = vzero;
    st0 = wmma_f16(kf.a00.f, qb0, st0);
    st0 = wmma_f16(kf.a01.f, qb1, st0);
    st1 = wmma_f16(kf.a10.f, qb0, st1);
    st1 = wmma_f16(kf.a11.f, qb1, st1);

    // V B-fragments (contiguous thanks to transposed Vt); issue early
    const _Float16* vc = vcol_l + j;
    v16h vb0 = __builtin_shufflevector(*(const v8h*)(vc), *(const v8h*)(vc + 16), 0, 1, 2, 3, 4, 5, 6, 7, 8, 9, 10, 11, 12, 13, 14, 15);
    v16h vb1 = __builtin_shufflevector(*(const v8h*)(vc + (size_t)16 * T), *(const v8h*)(vc + (size_t)16 * T + 16), 0, 1, 2, 3, 4, 5, 6, 7, 8, 9, 10, 11, 12, 13, 14, 15);
    v16h vb2 = __builtin_shufflevector(*(const v8h*)(vc + (size_t)32 * T), *(const v8h*)(vc + (size_t)32 * T + 16), 0, 1, 2, 3, 4, 5, 6, 7, 8, 9, 10, 11, 12, 13, 14, 15);
    v16h vb3 = __builtin_shufflevector(*(const v8h*)(vc + (size_t)48 * T), *(const v8h*)(vc + (size_t)48 * T + 16), 0, 1, 2, 3, 4, 5, 6, 7, 8, 9, 10, 11, 12, 13, 14, 15);
    __builtin_prefetch((const void*)(vc + 32), 0, 1);

    const v8f msk0 = *(const v8f*)(mrow_l + j);
    const v8f msk1 = *(const v8f*)(mrow_l + j + 16);

    const int kb = j + 8 * hi;
    float a0[8], a1[8];
#pragma unroll
    for (int v = 0; v < 8; ++v) {
      a0[v] = fmin_raw(st0[v], msk0[v]);   // masked -> exactly -1e9
      a1[v] = fmin_raw(st1[v], msk1[v]);
      if (causal) {
        a0[v] = ((kb + v) > srow) ? -1.0e9f : a0[v];
        a1[v] = ((kb + 16 + v) > srow) ? -1.0e9f : a1[v];
      }
    }

    // row max: per-lane tree + one xor-16 combine with the partner half
    float rm0 = fmax_raw(fmax_raw(fmax_raw(a0[0], a0[1]), fmax_raw(a0[2], a0[3])),
                         fmax_raw(fmax_raw(a0[4], a0[5]), fmax_raw(a0[6], a0[7])));
    float rm1 = fmax_raw(fmax_raw(fmax_raw(a1[0], a1[1]), fmax_raw(a1[2], a1[3])),
                         fmax_raw(fmax_raw(a1[4], a1[5]), fmax_raw(a1[6], a1[7])));
    float rm = fmax_raw(rm0, rm1);
    rm = fmax_raw(rm, __shfl_xor(rm, 16, 32));

    const float mn    = fmax_raw(m, rm);
    const float alpha = __builtin_exp2f(m - mn);   // native v_exp_f32
    const bool  chg   = mn > m;
    m = mn;

    // P = 2^(a - m): lane already holds the exact P A-fragment pattern
    FragA pa;
    float rs = 0.f;
#pragma unroll
    for (int v = 0; v < 8; ++v) {
      const float p0 = __builtin_exp2f(a0[v] - mn);
      const float p1 = __builtin_exp2f(a1[v] - mn);
      rs += p0 + p1;
      pa.h[0][v] = (_Float16)p0;
      pa.h[1][v] = (_Float16)p1;
    }
    rs += __shfl_xor(rs, 16, 32);
    l = l * alpha + rs;

    // rescale accumulators only if some row's max actually moved
    if (__ballot(chg)) {
#pragma unroll
      for (int v = 0; v < 8; ++v) {
        const float av = __shfl(alpha, v + 8 * hi, 32);
        acc0[v] *= av; acc1[v] *= av; acc2[v] *= av; acc3[v] *= av;
      }
    }

    acc0 = wmma_f16(pa.f, vb0, acc0);
    acc1 = wmma_f16(pa.f, vb1, acc1);
    acc2 = wmma_f16(pa.f, vb2, acc2);
    acc3 = wmma_f16(pa.f, vb3, acc3);
  };

  // Double-buffered pipeline: loads land directly in the buffer their tile
  // consumes (no rotating register copies).
  KF f0 = load_ka(0);
  int i = 0;
#pragma clang loop unroll(disable)
  for (; i + 2 < ntiles; i += 2) {
    KF f1 = load_ka((i + 1) << 5);
    tile(i << 5, f0, 0);
    f0 = load_ka((i + 2) << 5);
    tile((i + 1) << 5, f1, 0);
  }
  if (i + 2 == ntiles) {                 // two tiles left: main + causal tail
    KF f1 = load_ka((i + 1) << 5);
    tile(i << 5, f0, 0);
    tile((i + 1) << 5, f1, 1);
  } else {                               // one tile left: the causal tail
    tile(i << 5, f0, 1);
  }

  // epilogue: broadcast per-row l/m to the lanes holding that row's output
  const float* mvp = meanV + n * E + h * D;
#pragma unroll
  for (int v = 0; v < 8; ++v) {
    const int   r   = v + 8 * hi;
    const float lv  = __shfl(l, r, 32);
    const float mv  = __shfl(m, r, 32);
    const float inv = 1.0f / lv;
    const bool  deg = (mv == -1.0e9f);  // all in-range cols masked -> uniform
    float* ob = out + (size_t)(n * S + q0 + r) * E + h * D;
    ob[ln]      = deg ? mvp[ln]      : acc0[v] * inv;
    ob[16 + ln] = deg ? mvp[16 + ln] : acc1[v] * inv;
    ob[32 + ln] = deg ? mvp[32 + ln] : acc2[v] * inv;
    ob[48 + ln] = deg ? mvp[48 + ln] : acc3[v] * inv;
  }
}

// ---------------------------------------------------------------------------
extern "C" void kernel_launch(void* const* d_in, const int* in_sizes, int n_in,
                              void* d_out, int out_size, void* d_ws, size_t ws_size,
                              hipStream_t stream) {
  const float* q   = (const float*)d_in[0];
  const float* k   = (const float*)d_in[1];
  const float* v   = (const float*)d_in[2];
  const int*   pad = (const int*)d_in[3];
  // d_in[4] = subsq_mask (causal) -- computed analytically in-kernel
  const float* Wq  = (const float*)d_in[5];
  const float* bq  = (const float*)d_in[6];
  const float* Wk  = (const float*)d_in[7];
  const float* bk  = (const float*)d_in[8];
  const float* Wv  = (const float*)d_in[9];
  const float* bv  = (const float*)d_in[10];
  float* out = (float*)d_out;

  _Float16* Qp = (_Float16*)d_ws;            // [N,S,E] f16  (8 MB)
  _Float16* Kp = Qp + (size_t)R * E;         // [N,T,E] f16  (8 MB)
  _Float16* Vt = Kp + (size_t)R * E;         // [N,E,T] f16  (8 MB)
  float*    mV = (float*)(Vt + (size_t)R * E);   // [N,E]  f32 (8 KB)
  float*    mF = mV + (size_t)NB * E;            // [N,T]  f32 (16 KB)

  dim3 blk(256);
  const int proj_blocks = (R / 16) * (E / 16) / 8;  // 2048 blocks (8 waves ea.)
  proj_kernel<false><<<proj_blocks, blk, 0, stream>>>(q, Wq, bq, Qp, QSCALE);
  proj_kernel<false><<<proj_blocks, blk, 0, stream>>>(k, Wk, bk, Kp, 1.0f);
  proj_kernel<true><<<proj_blocks, blk, 0, stream>>>(v, Wv, bv, Vt, 1.0f);

  meanv_kernel<<<(NB * E) / 256, blk, 0, stream>>>(Vt, mV);
  padmask_kernel<<<(NB * T) / 256, blk, 0, stream>>>(pad, mF);

  const int attn_blocks = NB * H * (S / 16) / 8;    // 512 blocks
  attn_kernel<<<attn_blocks, blk, 0, stream>>>(Qp, Kp, Vt, mF, mV, out);
}
